// GATLayer_20461224198293
// MI455X (gfx1250) — matrix operand order repaired and checked by reading, not compiled, not run on hardware
//
#include <hip/hip_runtime.h>

#define NROWS 8192
#define FIN   512
#define FOUT  256
#define NSLOPE 0.2f

typedef __attribute__((ext_vector_type(16))) __bf16 v16bf;
typedef __attribute__((ext_vector_type(8)))  __bf16 v8bf;
typedef __attribute__((ext_vector_type(8)))  float  v8f;

#define SGB __builtin_amdgcn_sched_group_barrier
#define M_DSR  0x100   // DS read
#define M_WMMA 0x008   // MFMA/WMMA

__device__ __forceinline__ float lrelu(float z) { return fmaxf(z, NSLOPE * z); }

// ---------------- W -> bf16, transposed: WbT[n][k] = W[k][n] ----------------
__global__ void k_prep_w(const float* __restrict__ W, __bf16* __restrict__ WbT) {
    int idx = blockIdx.x * blockDim.x + threadIdx.x;
    if (idx >= FIN * FOUT) return;
    int k = idx / FOUT, n = idx % FOUT;
    WbT[(size_t)n * FIN + k] = (__bf16)W[(size_t)k * FOUT + n];
}

// ---------------- GEMM1: hT[n][r] = (X @ W)[r][n]  (bf16 WMMA) ----------------
__global__ void __launch_bounds__(256) k_gemm1(const float* __restrict__ X,
                                               const __bf16* __restrict__ WbT,
                                               __bf16* __restrict__ hT) {
    __shared__ __bf16 sX[128][32];   // A tile: rows x K
    __shared__ __bf16 sW[256][32];   // B tile: cols x K (K-contiguous)
    const int tid  = threadIdx.x;
    const int wave = tid >> 5, lane = tid & 31;
    const int lm   = lane & 15;
    const bool hi  = lane >= 16;
    const int wgrow = blockIdx.x * 128;
    const int ka = hi ? 16 : 0;

    v8f acc[16] = {};

    for (int k0 = 0; k0 < FIN; k0 += 32) {
        { // stage X (fp32 -> bf16): thread t covers row t/2, 16 K values
            int row = tid >> 1, halfk = (tid & 1) * 16;
            const float* src = X + (size_t)(wgrow + row) * FIN + k0 + halfk;
            #pragma unroll
            for (int e = 0; e < 16; ++e) sX[row][halfk + e] = (__bf16)src[e];
        }
        { // stage WbT row n: 64 contiguous bytes
            const uint4* src = (const uint4*)(WbT + (size_t)tid * FIN + k0);
            uint4* dst = (uint4*)&sW[tid][0];
            dst[0] = src[0]; dst[1] = src[1]; dst[2] = src[2]; dst[3] = src[3];
        }
        __syncthreads();

        const int arow = wave * 16 + lm;
        union { v16bf v; v8bf h[2]; } au;      // A frag = two b128 LDS reads
        au.h[0] = *(const v8bf*)&sX[arow][ka];
        au.h[1] = *(const v8bf*)&sX[arow][ka + 8];

        const int boff = hi ? 16 : 0;
        #pragma unroll
        for (int c = 0; c < 16; ++c) {
            v16bf b = *(const v16bf*)&sW[16 * c + lm][boff];
            acc[c] = __builtin_amdgcn_wmma_f32_16x16x32_bf16(
                false, au.v, false, b, (short)0, acc[c], false, false);
        }
        // Forced pipeline: A(2) | B0,B1(4) | {WMMA, next-B(2)}x14 | WMMA x2
        SGB(M_DSR, 2, 0);
        SGB(M_DSR, 4, 0);
        #pragma unroll
        for (int c = 0; c < 14; ++c) { SGB(M_WMMA, 1, 0); SGB(M_DSR, 2, 0); }
        SGB(M_WMMA, 2, 0);
        __syncthreads();
    }

    const int r0 = wgrow + wave * 16 + (hi ? 8 : 0);
    for (int c = 0; c < 16; ++c) {
        int col = 16 * c + lm;
        #pragma unroll
        for (int v = 0; v < 8; ++v)
            hT[(size_t)col * NROWS + r0 + v] = (__bf16)acc[c][v];
    }
}

// ---------------- s1_i = h_i . a1 ; s2_i = h_i . a2 ----------------
__global__ void k_stats(const __bf16* __restrict__ hT, const float* __restrict__ a,
                        float* __restrict__ s1, float* __restrict__ s2) {
    int i = blockIdx.x * blockDim.x + threadIdx.x;
    float acc1 = 0.f, acc2 = 0.f;
    for (int c = 0; c < FOUT; ++c) {
        float hv = (float)hT[(size_t)c * NROWS + i];   // coalesced in i
        acc1 += hv * a[c];
        acc2 += hv * a[FOUT + c];
    }
    s1[i] = acc1; s2[i] = acc2;
}

// ---------------- s2max (lrelu monotonic => m_i = lrelu(s1_i + s2max)) -------
__global__ void k_s2max(const float* __restrict__ s2, float* __restrict__ s2max) {
    __shared__ float red[1024];
    float m = -3.4e38f;
    for (int j = threadIdx.x; j < NROWS; j += 1024) m = fmaxf(m, s2[j]);
    red[threadIdx.x] = m; __syncthreads();
    for (int s = 512; s > 0; s >>= 1) {
        if (threadIdx.x < (unsigned)s)
            red[threadIdx.x] = fmaxf(red[threadIdx.x], red[threadIdx.x + s]);
        __syncthreads();
    }
    if (threadIdx.x == 0) *s2max = red[0];
}

// ---------------- l_i = sum_j exp(lrelu(s1_i+s2_j) - m_i); store 1/l ---------
__global__ void k_lsum(const float* __restrict__ s1, const float* __restrict__ s2,
                       const float* __restrict__ s2max, float* __restrict__ linv) {
    int i = blockIdx.x * blockDim.x + threadIdx.x;
    float s1i = s1[i];
    float m = lrelu(s1i + *s2max);
    float l = 0.f;
    for (int j = 0; j < NROWS; ++j)              // s2 broadcast, L2 resident
        l += __expf(lrelu(s1i + s2[j]) - m);
    linv[i] = 1.0f / l;
}

// -------- GEMM2 (flash-style): out = elu(P @ h), P built in-register --------
// Grid: 128 blocks = 64 row-groups (256 rows) x 2 col-groups (128 cols).
// Wave w: rows [wgrow + 32w, +32) (2 A-frags), all 128 cols (8 N-tiles).
// Each B fragment feeds 2 WMMAs; sched_group_barriers force the
// ds_load / v_wmma software pipeline. 1 barrier per k-step (double buffer).
__global__ void __launch_bounds__(256) k_gemm2(const __bf16* __restrict__ hT,
                                               const float* __restrict__ s1,
                                               const float* __restrict__ s2,
                                               const float* __restrict__ s2maxp,
                                               const float* __restrict__ linv,
                                               float* __restrict__ out) {
    __shared__ __bf16 sH[2][128][32];   // 2 x 8KB B tiles
    __shared__ float s2c[2][32];
    const int tid  = threadIdx.x;
    const int wave = tid >> 5, lane = tid & 31;
    const int lm   = lane & 15;
    const bool hi  = lane >= 16;
    const int wgrow  = (blockIdx.x >> 1) * 256;
    const int coloff = (blockIdx.x & 1) * 128;
    const int ka   = hi ? 16 : 0;
    const int boff = hi ? 16 : 0;

    const int arow0 = wgrow + wave * 32 + lm;   // A-frag 0 row (M = lm)
    const int arow1 = arow0 + 16;               // A-frag 1 row
    const float s2mx = *s2maxp;
    const float s1r0 = s1[arow0], s1r1 = s1[arow1];
    const float mr0 = lrelu(s1r0 + s2mx), mr1 = lrelu(s1r1 + s2mx);
    const float lr0 = linv[arow0],        lr1 = linv[arow1];

    v8f acc0[8] = {};
    v8f acc1[8] = {};

    // register staging: thread t stages 32B of row n = t/2, half = t%2
    const int srow = tid >> 1, shalf = (tid & 1) * 16;
    const __bf16* gsrc = hT + (size_t)(coloff + srow) * NROWS + shalf;
    uint4 st0 = ((const uint4*)(gsrc))[0];
    uint4 st1 = ((const uint4*)(gsrc))[1];
    float s2v = (tid < 32) ? s2[tid] : 0.f;

    int p = 0;
    for (int j0 = 0; j0 < NROWS; j0 += 32) {
        { // commit staged registers to LDS buffer p
            uint4* dst = (uint4*)&sH[p][srow][shalf];
            dst[0] = st0; dst[1] = st1;
        }
        if (tid < 32) s2c[p][tid] = s2v;
        __syncthreads();

        if (j0 + 32 < NROWS) {  // pre-stage next tile; latency hides in WMMAs
            const uint4* src = (const uint4*)(gsrc + j0 + 32);
            st0 = src[0]; st1 = src[1];
            if (tid < 32) s2v = s2[j0 + 32 + tid];
            if (j0 + 64 < NROWS)
                __builtin_prefetch(gsrc + j0 + 64, 0, 1);   // global_prefetch_b8
        }

        // Build 2 normalized attention A-fragments on the fly (1/l folded in).
        v8f z0v = *(const v8f*)&s2c[p][ka];          // 2x ds_load_b128
        v8f z1v = *(const v8f*)&s2c[p][ka + 8];      // 2x ds_load_b128
        v16bf a0, a1;
        #pragma unroll
        for (int e = 0; e < 8; ++e) {
            float z0 = z0v[e], z1 = z1v[e];
            a0[e]     = (__bf16)(__expf(lrelu(s1r0 + z0) - mr0) * lr0);
            a0[8 + e] = (__bf16)(__expf(lrelu(s1r0 + z1) - mr0) * lr0);
            a1[e]     = (__bf16)(__expf(lrelu(s1r1 + z0) - mr1) * lr1);
            a1[8 + e] = (__bf16)(__expf(lrelu(s1r1 + z1) - mr1) * lr1);
        }

        #pragma unroll
        for (int c = 0; c < 8; ++c) {
            v16bf b = *(const v16bf*)&sH[p][16 * c + lm][boff];
            acc0[c] = __builtin_amdgcn_wmma_f32_16x16x32_bf16(
                false, a0, false, b, (short)0, acc0[c], false, false);
            acc1[c] = __builtin_amdgcn_wmma_f32_16x16x32_bf16(
                false, a1, false, b, (short)0, acc1[c], false, false);
        }
        // Forced pipeline: z(4) | B0,B1(4) | {WMMAx2, next-B(2)}x6 | WMMAx4
        SGB(M_DSR, 4, 0);
        SGB(M_DSR, 4, 0);
        #pragma unroll
        for (int c = 0; c < 6; ++c) { SGB(M_WMMA, 2, 0); SGB(M_DSR, 2, 0); }
        SGB(M_WMMA, 4, 0);
        p ^= 1;
    }

    // Epilogue: ELU, write row-major fp32 (lanes 0-15 -> 16 consecutive cols)
    #pragma unroll
    for (int mt = 0; mt < 2; ++mt) {
        const int r0 = wgrow + wave * 32 + mt * 16 + (hi ? 8 : 0);
        for (int c = 0; c < 8; ++c) {
            int col = coloff + 16 * c + lm;
            #pragma unroll
            for (int v = 0; v < 8; ++v) {
                float x = mt ? acc1[c][v] : acc0[c][v];
                out[(size_t)(r0 + v) * FOUT + col] = x > 0.f ? x : (__expf(x) - 1.0f);
            }
        }
    }
}

// ---------------- host side ----------------
extern "C" void kernel_launch(void* const* d_in, const int* in_sizes, int n_in,
                              void* d_out, int out_size, void* d_ws, size_t ws_size,
                              hipStream_t stream) {
    const float* X = (const float*)d_in[0];   // [8192,512]
    const float* W = (const float*)d_in[1];   // [512,256]
    const float* a = (const float*)d_in[2];   // [512,1]
    float* out = (float*)d_out;               // [8192,256]

    char* ws = (char*)d_ws;
    __bf16* hT  = (__bf16*)ws;                                   // 4 MB
    __bf16* WbT = (__bf16*)(ws + (size_t)FOUT * NROWS * 2);      // 256 KB
    float* s1   = (float*)(ws + (size_t)FOUT * NROWS * 2 + (size_t)FOUT * FIN * 2);
    float* s2    = s1 + NROWS;
    float* linv  = s2 + NROWS;
    float* s2max = linv + NROWS;

    k_prep_w<<<(FIN * FOUT + 255) / 256, 256, 0, stream>>>(W, WbT);
    k_gemm1 <<<NROWS / 128, 256, 0, stream>>>(X, WbT, hT);
    k_stats <<<NROWS / 256, 256, 0, stream>>>(hT, a, s1, s2);
    k_s2max <<<1, 1024, 0, stream>>>(s2, s2max);
    k_lsum  <<<NROWS / 256, 256, 0, stream>>>(s1, s2, s2max, linv);
    k_gemm2 <<<NROWS / 64, 256, 0, stream>>>(hT, s1, s2, s2max, linv, out);
}
